// Relaxed_EquivResBlock_82188494176372
// MI455X (gfx1250) — matrix operand order repaired and checked by reading, not compiled, not run on hardware
//
#include <hip/hip_runtime.h>
#include <stdint.h>

typedef __attribute__((ext_vector_type(16))) __bf16 v16bf;
typedef __attribute__((ext_vector_type(8)))  float  v8f;
typedef uint32_t u32x4 __attribute__((ext_vector_type(4)));
typedef uint32_t u32x8 __attribute__((ext_vector_type(8)));

#define CI   384
#define CO   384
#define SPV  32768              // 32^3 voxels per batch sample
#define KDIM (CI * 27)          // 10368, ordered k = tap*384 + ic
#define NB   2
#define BM   128
#define BN   128
#define BK   32
#define NCHUNK (KDIM / BK)      // 324
#define LDA_STRIDE 40           // bf16 elems per A row in LDS (64B data + 16B pad)
#define LDB_STRIDE 40
#define PDIM 34                 // padded spatial edge
#define PSLICE (PDIM * PDIM)    // 1156
#define PVOL  (PDIM * PDIM * PDIM) // 39304

// ---------------------------------------------------------------------------
// 1) Octahedral group tables (SHIFT 24x24, ROT 24x27), exact itertools order.
// ---------------------------------------------------------------------------
__global__ void octa_tables_kernel(int* __restrict__ shift, int* __restrict__ rot) {
    if (threadIdx.x != 0 || blockIdx.x != 0) return;
    const int perms[6][3] = {{0,1,2},{0,2,1},{1,0,2},{1,2,0},{2,0,1},{2,1,0}};
    int M[24][3][3];
    int cnt = 0;
    for (int pi = 0; pi < 6; ++pi) {
        for (int si = 0; si < 8; ++si) {
            int sg[3] = { (si & 4) ? -1 : 1, (si & 2) ? -1 : 1, (si & 1) ? -1 : 1 };
            int T[3][3] = {{0,0,0},{0,0,0},{0,0,0}};
            for (int i = 0; i < 3; ++i) T[i][perms[pi][i]] = sg[i];
            int det = T[0][0]*(T[1][1]*T[2][2]-T[1][2]*T[2][1])
                    - T[0][1]*(T[1][0]*T[2][2]-T[1][2]*T[2][0])
                    + T[0][2]*(T[1][0]*T[2][1]-T[1][1]*T[2][0]);
            if (det == 1) {
                for (int i = 0; i < 3; ++i)
                    for (int j = 0; j < 3; ++j) M[cnt][i][j] = T[i][j];
                ++cnt;
            }
        }
    }
    for (int g = 0; g < 24; ++g) {
        for (int h = 0; h < 24; ++h) {
            int P[3][3];
            for (int i = 0; i < 3; ++i)
                for (int j = 0; j < 3; ++j) {
                    int s = 0;
                    for (int k = 0; k < 3; ++k) s += M[g][k][i] * M[h][k][j];
                    P[i][j] = s;
                }
            int idx = -1;
            for (int m = 0; m < 24 && idx < 0; ++m) {
                bool eq = true;
                for (int i = 0; i < 3; ++i)
                    for (int j = 0; j < 3; ++j)
                        if (P[i][j] != M[m][i][j]) eq = false;
                if (eq) idx = m;
            }
            shift[g * 24 + h] = idx;
        }
    }
    for (int g = 0; g < 24; ++g) {
        for (int p = 0; p < 27; ++p) {
            int c[3] = { p / 9, (p / 3) % 3, p % 3 };
            int s[3];
            for (int j = 0; j < 3; ++j) {
                int acc = 0;
                for (int i = 0; i < 3; ++i) acc += (c[i] - 1) * M[g][i][j];
                s[j] = acc + 1;
            }
            rot[g * 27 + p] = s[0] * 9 + s[1] * 3 + s[2];
        }
    }
}

// ---------------------------------------------------------------------------
// 2) Weight transform -> bf16 GEMM-A layout [oc][tap*384 + ic] (tap-major K)
// ---------------------------------------------------------------------------
__global__ __launch_bounds__(256)
void weight_transform_kernel(const float* __restrict__ w,   // [NB][16][16][24][27]
                             const float* __restrict__ rw,  // [NB][24]
                             const int* __restrict__ shift,
                             const int* __restrict__ rot,
                             __bf16* __restrict__ cwb) {    // [384][KDIM]
    const int TOT = CO * KDIM;
    int idx = blockIdx.x * blockDim.x + threadIdx.x;
    if (idx >= TOT) return;
    int ic = idx % CI;
    int p  = (idx / CI) % 27;
    int oc = idx / KDIM;
    int o = oc / 24, g = oc % 24;
    int i = ic / 24, h = ic % 24;
    int gp = shift[g * 24 + h];
    int rp = rot[g * 27 + p];
    float acc = 0.f;
#pragma unroll
    for (int n = 0; n < NB; ++n)
        acc = fmaf(rw[n * 24 + g], w[(((size_t)(n * 16 + o) * 16 + i) * 24 + gp) * 27 + rp], acc);
    cwb[idx] = (__bf16)acc;
}

// ---------------------------------------------------------------------------
// 3) Pad+convert f32 -> bf16 halo layout [768][34][34][34];  4) zero-fill
// ---------------------------------------------------------------------------
__global__ __launch_bounds__(256)
void pad_bf16_kernel(const float* __restrict__ in, __bf16* __restrict__ out, int total) {
    int idx = blockIdx.x * blockDim.x + threadIdx.x;
    if (idx >= total) return;
    int c = idx / PVOL;
    int r = idx - c * PVOL;
    int z = r / PSLICE;
    int y = (r / PDIM) % PDIM;
    int x = r % PDIM;
    float v = 0.f;
    if (z >= 1 && z <= 32 && y >= 1 && y <= 32 && x >= 1 && x <= 32)
        v = in[((size_t)c << 15) + ((z - 1) << 10) + ((y - 1) << 5) + (x - 1)];
    out[idx] = (__bf16)v;
}

__global__ __launch_bounds__(256)
void zero_bf16_kernel(__bf16* __restrict__ out, int total) {
    int idx = blockIdx.x * blockDim.x + threadIdx.x;
    if (idx < total) out[idx] = (__bf16)0.f;
}

// ---------------------------------------------------------------------------
// 5) Implicit-GEMM conv: tap-major K, double-buffered LDS, TDM-staged A tile,
//    bf16 WMMA (8 per wave per K-chunk).
//    EPILOGUE 0: h = leaky(conv + b)        -> bf16 into padded layout
//    EPILOGUE 1: y = leaky(conv + b + skip) -> f32 linear
// ---------------------------------------------------------------------------
template <int EPILOGUE>
__global__ __launch_bounds__(256)
void conv_wmma_kernel(const __bf16* __restrict__ xin,   // padded [768][34][34][34]
                      const __bf16* __restrict__ wt,    // [384][KDIM]
                      const float*  __restrict__ bias,  // [16]
                      const float*  __restrict__ skip,  // f32 linear (EPILOGUE==1)
                      float* __restrict__ out_f32,
                      __bf16* __restrict__ out_pad) {
    __shared__ __align__(16) __bf16 Asmem[2][BM * LDA_STRIDE];
    __shared__ __align__(16) __bf16 Bsmem[2][BN * LDB_STRIDE];

    const int t    = threadIdx.x;
    const int lane = t & 31;
    const int wave = t >> 5;
    const int wm   = wave >> 1;              // 0..3 -> M offset wm*32
    const int wn   = wave & 1;               // 0..1 -> N offset wn*64
    const int mTile = blockIdx.x % (CO / BM);
    const int nTile = blockIdx.x / (CO / BM);
    const int ocBase  = mTile * BM;
    const int colBase = nTile * BN;

    // ---- B-gather geometry: thread owns 1 spatial column x 16 channels
    const int nRow   = t >> 1;               // 0..127
    const int icHalf = (t & 1) * 16;         // 0 or 16
    const int j0 = colBase + nRow;
    const int bIdx = j0 >> 15;
    const int sIdx = j0 & (SPV - 1);
    const int z = sIdx >> 10, y = (sIdx >> 5) & 31, x0 = sIdx & 31;
    const size_t chanBase = (size_t)bIdx * CI * PVOL;
    const int    sBase    = (z + 1) * PSLICE + (y + 1) * PDIM + (x0 + 1);

    // ---- TDM D# group 1 (constant) and per-buffer LDS addresses
    const uint32_t ldsA0 = (uint32_t)(uintptr_t)&Asmem[0][0];
    const uint32_t ldsA1 = (uint32_t)(uintptr_t)&Asmem[1][0];
    u32x8 g1;
    g1[0] = (1u << 16)            // data_size = 2 bytes
          | (1u << 20)            // pad_enable
          | (3u << 22)            // pad_interval: 16 dwords (= one 64B row)
          | (3u << 25);           // pad_amount: 4 dwords (-> 80B LDS row stride)
    g1[1] = ((uint32_t)KDIM & 0xFFFFu) << 16;              // tensor_dim0 lo
    g1[2] = ((uint32_t)KDIM >> 16) | ((uint32_t)CO << 16); // dim0 hi | dim1 lo
    g1[3] = (32u << 16);                                   // tile_dim0 = 32 elems
    g1[4] = 128u;                                          // tile_dim1 = 128 rows
    g1[5] = (uint32_t)KDIM;                                // tensor_dim0_stride lo
    g1[6] = 0u;
    g1[7] = 0u;

    // ---- stage one K-chunk (tap, icb) into LDS buffer `buf`
    auto stage = [&](int tap, int icb, int buf) {
        if (wave == 0) {
            const int k0 = tap * CI + icb;
            uint64_t ga = (uint64_t)(uintptr_t)(wt + (size_t)ocBase * KDIM + k0);
            u32x4 g0;
            g0[0] = 1u;                                  // count = 1 (valid D#)
            g0[1] = buf ? ldsA1 : ldsA0;                 // lds_addr
            g0[2] = (uint32_t)ga;                        // global_addr[31:0]
            g0[3] = (uint32_t)((ga >> 32) & 0x1FFFFFFu)  // global_addr[56:32]
                  | (2u << 30);                          // type = 2 ("image")
            asm volatile("tensor_load_to_lds %0, %1" :: "s"(g0), "s"(g1) : "memory");
        }
        const int dz = tap / 9 - 1, dy = (tap / 3) % 3 - 1, dx = tap % 3 - 1;
        const int tapOff = dz * PSLICE + dy * PDIM + dx;
        const __bf16* src = xin + chanBase + (size_t)(icb + icHalf) * PVOL
                          + (sBase + tapOff);
        union { __bf16 h[16]; uint4 u[2]; } gb;
#pragma unroll
        for (int q = 0; q < 16; ++q) gb.h[q] = src[(size_t)q * PVOL];
        uint4* dst = (uint4*)&Bsmem[buf][nRow * LDB_STRIDE + icHalf];
        dst[0] = gb.u[0];
        dst[1] = gb.u[1];
    };

    v8f acc[2][4];
#pragma unroll
    for (int i = 0; i < 2; ++i)
#pragma unroll
        for (int j = 0; j < 4; ++j) acc[i][j] = (v8f){};

    // ---- prologue: stage chunk 0 into buffer 0
    stage(0, 0, 0);
    if (wave == 0) __builtin_amdgcn_s_wait_tensorcnt(0);
    __syncthreads();

    int ntap = 0, nicb = BK;                 // next chunk to stage (= chunk 1)
    if (nicb == CI) { nicb = 0; ++ntap; }

    for (int kc = 0; kc < NCHUNK; ++kc) {
        const int cur  = kc & 1;
        const bool more = (kc + 1 < NCHUNK);

        // ---- stage next chunk into the other buffer (overlaps with compute)
        if (more) stage(ntap, nicb, cur ^ 1);

        // ---- fragments from current buffer (CDNA5 16-bit WMMA lane layouts)
        union Frag { v16bf bf; uint4 u[2]; };
        Frag fa[2], fb[4];
        {
            const int half = lane >> 4;
#pragma unroll
            for (int i = 0; i < 2; ++i) {
                int m = wm * 32 + i * 16 + (lane & 15);
                const __bf16* a = &Asmem[cur][m * LDA_STRIDE + half * 8];
                fa[i].u[0] = *(const uint4*)(a);       // k = 8*half..+7
                fa[i].u[1] = *(const uint4*)(a + 16);  // k = 16+8*half..+7
            }
#pragma unroll
            for (int j = 0; j < 4; ++j) {
                int n = wn * 64 + j * 16 + (lane & 15);
                const __bf16* b = &Bsmem[cur][n * LDB_STRIDE + half * 8];
                fb[j].u[0] = *(const uint4*)(b);       // k = 8*half..+7
                fb[j].u[1] = *(const uint4*)(b + 16);  // k = 16+8*half..+7
            }
        }
#pragma unroll
        for (int i = 0; i < 2; ++i)
#pragma unroll
            for (int j = 0; j < 4; ++j)
                acc[i][j] = __builtin_amdgcn_wmma_f32_16x16x32_bf16(
                    false, fa[i].bf, false, fb[j].bf, (short)0, acc[i][j],
                    false, false);

        // ---- publish next buffer; wave0 first drains its TDM
        if (more && wave == 0) __builtin_amdgcn_s_wait_tensorcnt(0);
        __syncthreads();
        if (more) {
            nicb += BK;
            if (nicb == CI) { nicb = 0; ++ntap; }
        }
    }

    // ---- epilogue: C layout -> m = r + 8*(lane>>4), n = lane&15
    const int nF   = lane & 15;
    const int mOff = (lane >> 4) * 8;
#pragma unroll
    for (int i = 0; i < 2; ++i) {
#pragma unroll
        for (int r = 0; r < 8; ++r) {
            int oc   = ocBase + wm * 32 + i * 16 + mOff + r;
            float bv = bias[oc / 24];
#pragma unroll
            for (int j = 0; j < 4; ++j) {
                int col = colBase + wn * 64 + j * 16 + nF;
                int b   = col >> 15;
                int s   = col & (SPV - 1);
                float v = acc[i][j][r] + bv;
                if (EPILOGUE == 0) {
                    v = v > 0.f ? v : 0.01f * v;
                    int zz = s >> 10, yy = (s >> 5) & 31, xx = s & 31;
                    size_t pidx = (size_t)(b * CO + oc) * PVOL
                                + (zz + 1) * PSLICE + (yy + 1) * PDIM + (xx + 1);
                    out_pad[pidx] = (__bf16)v;
                } else {
                    size_t gidx = ((size_t)(b * CO + oc) << 15) + s;
                    v += skip[gidx];
                    v = v > 0.f ? v : 0.01f * v;
                    out_f32[gidx] = v;
                }
            }
        }
    }
}

// ---------------------------------------------------------------------------
// Launch
// ---------------------------------------------------------------------------
extern "C" void kernel_launch(void* const* d_in, const int* in_sizes, int n_in,
                              void* d_out, int out_size, void* d_ws, size_t ws_size,
                              hipStream_t stream) {
    const float* x   = (const float*)d_in[0];
    const float* w1  = (const float*)d_in[1];
    const float* rw1 = (const float*)d_in[2];
    const float* b1  = (const float*)d_in[3];
    const float* w2  = (const float*)d_in[4];
    const float* rw2 = (const float*)d_in[5];
    const float* b2  = (const float*)d_in[6];
    float* out = (float*)d_out;

    char* ws = (char*)d_ws;
    size_t cur = 0;
    auto carve = [&](size_t bytes) {
        size_t p = cur;
        cur = (cur + bytes + 255) & ~(size_t)255;
        return p;
    };
    const int NPAD = 2 * CI * PVOL;                       // 30,185,472 elements
    int*    shift = (int*)(ws + carve(576 * sizeof(int)));
    int*    rot   = (int*)(ws + carve(648 * sizeof(int)));
    __bf16* cwb1  = (__bf16*)(ws + carve((size_t)CO * KDIM * 2));
    __bf16* cwb2  = (__bf16*)(ws + carve((size_t)CO * KDIM * 2));
    __bf16* xpad  = (__bf16*)(ws + carve((size_t)NPAD * 2));
    __bf16* hpad  = (__bf16*)(ws + carve((size_t)NPAD * 2));

    octa_tables_kernel<<<1, 1, 0, stream>>>(shift, rot);

    pad_bf16_kernel<<<(NPAD + 255) / 256, 256, 0, stream>>>(x, xpad, NPAD);
    zero_bf16_kernel<<<(NPAD + 255) / 256, 256, 0, stream>>>(hpad, NPAD);

    const int WTOT = CO * KDIM;
    weight_transform_kernel<<<(WTOT + 255) / 256, 256, 0, stream>>>(w1, rw1, shift, rot, cwb1);
    weight_transform_kernel<<<(WTOT + 255) / 256, 256, 0, stream>>>(w2, rw2, shift, rot, cwb2);

    const int nBlocks = (CO / BM) * ((2 * SPV) / BN);     // 3 * 512 = 1536
    conv_wmma_kernel<0><<<nBlocks, 256, 0, stream>>>(xpad, cwb1, b1, nullptr, nullptr, hpad);
    conv_wmma_kernel<1><<<nBlocks, 256, 0, stream>>>(hpad, cwb2, b2, x, out, nullptr);
}
